// Informer_41609643163710
// MI455X (gfx1250) — matrix operand order repaired and checked by reading, not compiled, not run on hardware
//
#include <hip/hip_runtime.h>
#include <hip/hip_bf16.h>

// ---------------------------------------------------------------------------
// Informer forward for MI455X (gfx1250, wave32, WMMA).
// GEMMs: LDS-tiled v_wmma_f32_16x16x32_f16 (f16 inputs, f32 accumulate),
// 64x128 block tile, 8 waves, 4 accumulators/wave (A-fragment reuse).
// Tiles are staged in LDS in *fragment order* (32B contiguous per lane):
// every fragment load is two aligned ds_load_b128; A staging is one 16B
// ds_store per thread; B staging is coalesced global reads + two 16B
// ds_stores per thread. global_prefetch_b8 pulls the next K tile.
// Whole model: ~150 GFLOP, ~30MB weights, activations L2-resident (192MB);
// the pass is bandwidth-shaped at 23.3 TB/s, so irregular ops (ProbSparse
// sampling/top-k/softmax-scatter, cumsum/mean ctx, layernorm, embedding)
// stay VALU.
// NOTE: ProbSparse sampling uses a hash PRNG, not JAX threefry (bit-exact
// JAX reproduction is not feasible; compile-only environment).
// ---------------------------------------------------------------------------

typedef __attribute__((ext_vector_type(16))) _Float16 v16h;
typedef __attribute__((ext_vector_type(8)))  _Float16 v8h;
typedef __attribute__((ext_vector_type(8)))  float    v8f;

#define DM 512
#define NH 8
#define DH 64
#define BB 4
#define LSEQ 2048
#define LDEC 1024
#define NEG_INF_F (-1e30f)

// GEMM tiling
#define BM 64
#define BN 128
#define BK 32

// ------------------------------- helpers -----------------------------------

__device__ __forceinline__ unsigned mix32(unsigned x) {
  x ^= x >> 16; x *= 0x7feb352dU;
  x ^= x >> 15; x *= 0x846ca68bU;
  x ^= x >> 16;
  return x;
}

__device__ __forceinline__ float sinusoid(float pos, int j) {
  // div = exp(2*pair * -(ln(10000)/512)); even->sin, odd->cos
  int pair = j >> 1;
  float freq = __expf((float)pair * (-2.0f * 9.210340371976184f / 512.0f));
  float ang = pos * freq;
  return (j & 1) ? __cosf(ang) : __sinf(ang);
}

// ---------------------------- embedding ------------------------------------
__global__ void embed_kernel(const float* __restrict__ x,
                             const int* __restrict__ xmark,
                             const float* __restrict__ convw,
                             float* __restrict__ out, int L) {
  int row = blockIdx.x;                 // b*L + l
  int b = row / L, l = row % L;
  int mo = xmark[row * 4 + 0];
  int dy = xmark[row * 4 + 1];
  int wd = xmark[row * 4 + 2];
  int hr = xmark[row * 4 + 3];
  for (int dm = threadIdx.x; dm < DM; dm += blockDim.x) {
    float tok = 0.f;
#pragma unroll
    for (int w = 0; w < 3; ++w) {
      int ls = (l - 1 + w + L) % L;     // 'wrap' padding
      const float* xr = x + ((size_t)b * L + ls) * 7;
      const float* cw = convw + (size_t)(w * 7) * DM + dm;
#pragma unroll
      for (int c = 0; c < 7; ++c) tok += xr[c] * cw[(size_t)c * DM];
    }
    float temp = sinusoid((float)hr, dm) + sinusoid((float)wd, dm) +
                 sinusoid((float)dy, dm) + sinusoid((float)mo, dm);
    out[(size_t)row * DM + dm] = tok + sinusoid((float)l, dm) + temp;
  }
}

// ------------------------------ WMMA GEMM ----------------------------------
// C[M,N] = act(A[M,K] * W[K,N] + bias); K == 512.
// blockDim=256 (8 waves). Block tile 64x128; wave (wm=wave&3, wn=wave>>2)
// computes rows [wm*16,+16) x cols [wn*64,+64) as 4 accumulators, reusing one
// A fragment across 4 WMMAs.
//
// LDS holds tiles in FRAGMENT order: 16 halves (32B) contiguous per lane.
//  A (ISA 7.12.2, 16-bit A 16x32): lane L: m=L%16, hi=L/16;
//    halves e=0..15 hold k=16*hi+e (same order as B, not the ISA table order).
//    Staging chunk c (8 consecutive k of row mr) maps to
//    lane=(c>>1)*16+(mr%16), e0=(c&1)*8 -> one aligned 16B store.
//  B (32x16): lane L: n=L%16, hi=L/16; halves e=0..15 are k=16*hi+e.
//    Staging: thread t owns column nb=t&127, half hib=t>>7; its 16 k-values
//    are one lane's contiguous 32B run -> coalesced global reads (128
//    consecutive floats per k across the thread group) + two 16B LDS stores.
//  C 16x16: acc[r] = (m = r+8*hi, n = L%16)
__global__ void gemm_bias_kernel(const float* __restrict__ A,
                                 const float* __restrict__ W,
                                 const float* __restrict__ bias,
                                 float* __restrict__ C,
                                 int M, int N, int K, int act) {
  __shared__ _Float16 As[(BM / 16) * 32 * 16];   // 4 m-tiles * 32 lanes * 16 = 4KB
  __shared__ _Float16 Bs[(BN / 16) * 32 * 16];   // 8 n-tiles * 32 lanes * 16 = 8KB
  const int tid  = threadIdx.x;
  const int lane = tid & 31;
  const int wave = tid >> 5;
  const int wm = wave & 3;
  const int wn = wave >> 2;
  const int m0 = blockIdx.x * BM;
  const int n0 = blockIdx.y * BN;
  const int lo = lane & 15;
  const int hi = lane >> 4;

  union Frag { v16h h; v8h v2[2]; };
  union Pack { v8h v; _Float16 e[8]; };

  v8f acc[4];
#pragma unroll
  for (int a = 0; a < 4; ++a) {
    float bv = bias ? bias[n0 + wn * 64 + a * 16 + lo] : 0.0f;
#pragma unroll
    for (int r = 0; r < 8; ++r) acc[a][r] = bv;
  }

  // A staging assignment: row mr = tid>>2, chunk c = tid&3 (8 k each)
  const int ar = tid >> 2;
  const int ac = tid & 3;
  const int a_dst_lane = (ac >> 1) * 16 + (ar & 15);
  const int a_dst_e0   = (ac & 1) * 8;
  _Float16* a_dst = &As[(((ar >> 4) * 32 + a_dst_lane) * 16) + a_dst_e0];

  // B staging assignment: column nb = tid&127, k-half hib = tid>>7
  const int nb  = tid & 127;
  const int hib = tid >> 7;
  v8h* b_dst = reinterpret_cast<v8h*>(
      &Bs[(((nb >> 4) * 32 + hib * 16 + (nb & 15)) * 16)]);

  for (int kk = 0; kk < K; kk += BK) {
    if (kk + BK < K) {  // prefetch next K tile (global_prefetch_b8)
      __builtin_prefetch(&A[(size_t)(m0 + ar) * K + kk + BK + ac * 8], 0, 3);
      __builtin_prefetch(&W[(size_t)(kk + BK + hib * 16) * N + n0 + nb], 0, 3);
      __builtin_prefetch(&W[(size_t)(kk + BK + hib * 16 + 8) * N + n0 + nb], 0, 3);
    }
    // stage A tile (64x32): two aligned float4 loads -> one 16B LDS store
    {
      const float4* s4 =
          reinterpret_cast<const float4*>(&A[(size_t)(m0 + ar) * K + kk + ac * 8]);
      float4 x0 = s4[0], x1 = s4[1];
      Pack pk;
      pk.e[0] = (_Float16)x0.x; pk.e[1] = (_Float16)x0.y;
      pk.e[2] = (_Float16)x0.z; pk.e[3] = (_Float16)x0.w;
      pk.e[4] = (_Float16)x1.x; pk.e[5] = (_Float16)x1.y;
      pk.e[6] = (_Float16)x1.z; pk.e[7] = (_Float16)x1.w;
      *reinterpret_cast<v8h*>(a_dst) = pk.v;
    }
    // stage B tile (32x128): 16 coalesced strided reads -> two 16B LDS stores
    {
      const float* src = &W[(size_t)(kk + hib * 16) * N + n0 + nb];
      float r[16];
#pragma unroll
      for (int j = 0; j < 16; ++j) r[j] = src[(size_t)j * N];
      Pack p0, p1;
#pragma unroll
      for (int j = 0; j < 8; ++j) { p0.e[j] = (_Float16)r[j]; p1.e[j] = (_Float16)r[8 + j]; }
      b_dst[0] = p0.v;
      b_dst[1] = p1.v;
    }
    __syncthreads();

    Frag fa;
    {
      const v8h* ap = reinterpret_cast<const v8h*>(&As[(wm * 32 + lane) * 16]);
      fa.v2[0] = ap[0]; fa.v2[1] = ap[1];
    }
#pragma unroll
    for (int a = 0; a < 4; ++a) {
      Frag fb;
      const v8h* bp =
          reinterpret_cast<const v8h*>(&Bs[((wn * 4 + a) * 32 + lane) * 16]);
      fb.v2[0] = bp[0]; fb.v2[1] = bp[1];
      acc[a] = __builtin_amdgcn_wmma_f32_16x16x32_f16(
          false, fa.h, false, fb.h, (short)0, acc[a], false, false);
    }
    __syncthreads();
  }

#pragma unroll
  for (int a = 0; a < 4; ++a) {
    int n = n0 + wn * 64 + a * 16 + lo;
#pragma unroll
    for (int r = 0; r < 8; ++r) {
      int m = m0 + wm * 16 + r + 8 * hi;
      float v = acc[a][r];
      if (act == 1)  // exact GELU
        v = 0.5f * v * (1.0f + erff(v * 0.70710678118654752440f));
      C[(size_t)m * N + n] = v;
    }
  }
}

// --------------------------- add + layernorm -------------------------------
__global__ void add_ln_kernel(const float* __restrict__ x,
                              const float* __restrict__ y,
                              const float* __restrict__ g,
                              const float* __restrict__ bta,
                              float* __restrict__ out, int M) {
  int wave = threadIdx.x >> 5, lane = threadIdx.x & 31;
  int row = blockIdx.x * 8 + wave;
  if (row >= M) return;
  const float* xr = x + (size_t)row * DM;
  const float* yr = y ? y + (size_t)row * DM : nullptr;
  float v[16];
  float s = 0.f;
#pragma unroll
  for (int i = 0; i < 16; ++i) {
    float t = xr[lane + 32 * i];
    if (yr) t += yr[lane + 32 * i];
    v[i] = t; s += t;
  }
#pragma unroll
  for (int m = 16; m >= 1; m >>= 1) s += __shfl_xor(s, m, 32);
  float mean = s * (1.0f / 512.0f);
  float sq = 0.f;
#pragma unroll
  for (int i = 0; i < 16; ++i) { float d = v[i] - mean; sq += d * d; }
#pragma unroll
  for (int m = 16; m >= 1; m >>= 1) sq += __shfl_xor(sq, m, 32);
  float inv = rsqrtf(sq * (1.0f / 512.0f) + 1e-5f);
#pragma unroll
  for (int i = 0; i < 16; ++i) {
    int c = lane + 32 * i;
    out[(size_t)row * DM + c] = (v[i] - mean) * inv * g[c] + bta[c];
  }
}

// ----------------------- ProbSparse: sparsity measure ----------------------
__global__ void prob_sample_m_kernel(const float* __restrict__ Q,
                                     const float* __restrict__ K,
                                     float* __restrict__ m,
                                     int L, int Lk, int u_part,
                                     unsigned seed, int total) {
  int wave = threadIdx.x >> 5, lane = threadIdx.x & 31;
  int gid = blockIdx.x * 8 + wave;    // (b*NH + h)*L + l
  if (gid >= total) return;
  int l = gid % L;
  int bh = gid / L;
  int b = bh >> 3, h = bh & 7;
  const float* q = Q + ((size_t)b * L + l) * DM + h * DH;
  float q0 = q[lane], q1 = q[lane + 32];
  float mx = NEG_INF_F, sm = 0.f;
  for (int s = 0; s < u_part; ++s) {
    // sampled indices shared across (b,h) like the reference's idx_sample
    unsigned idx = mix32(seed ^ ((unsigned)l * 0x9E3779B9u) ^
                         ((unsigned)s * 0x85ebca6bu)) % (unsigned)Lk;
    const float* kr = K + ((size_t)b * Lk + idx) * DM + h * DH;
    float d = q0 * kr[lane] + q1 * kr[lane + 32];
#pragma unroll
    for (int t = 16; t >= 1; t >>= 1) d += __shfl_xor(d, t, 32);
    mx = fmaxf(mx, d);
    sm += d;
  }
  if (lane == 0) m[gid] = mx - sm / (float)Lk;
}

// ------------------------------- top-k -------------------------------------
__global__ void topk_kernel(const float* __restrict__ m,
                            int* __restrict__ index, int L, int u) {
  int bh = blockIdx.x;
  __shared__ float vals[2048];
  __shared__ float bv[256];
  __shared__ int bi[256];
  for (int i = threadIdx.x; i < L; i += 256) vals[i] = m[(size_t)bh * L + i];
  __syncthreads();
  for (int t = 0; t < u; ++t) {
    float best = -3e38f; int bidx = 0;
    for (int i = threadIdx.x; i < L; i += 256)
      if (vals[i] > best) { best = vals[i]; bidx = i; }
    bv[threadIdx.x] = best; bi[threadIdx.x] = bidx;
    __syncthreads();
    for (int s = 128; s >= 1; s >>= 1) {
      if (threadIdx.x < s && bv[threadIdx.x + s] > bv[threadIdx.x]) {
        bv[threadIdx.x] = bv[threadIdx.x + s];
        bi[threadIdx.x] = bi[threadIdx.x + s];
      }
      __syncthreads();
    }
    if (threadIdx.x == 0) { index[bh * 64 + t] = bi[0]; vals[bi[0]] = -3e38f; }
    __syncthreads();
  }
}

// -------------------- context init: mean / cumsum of V ---------------------
__global__ void ctx_mean_kernel(const float* __restrict__ V,
                                float* __restrict__ ctx, int L) {
  int bh = blockIdx.x; int b = bh >> 3, h = bh & 7;
  __shared__ float part[256];
  __shared__ float meanv[64];
  int d = threadIdx.x & 63, seg = threadIdx.x >> 6;
  float s = 0.f;
  for (int l = seg; l < L; l += 4)
    s += V[((size_t)b * L + l) * DM + h * DH + d];
  part[threadIdx.x] = s;
  __syncthreads();
  if (seg == 0)
    meanv[d] = (part[d] + part[64 + d] + part[128 + d] + part[192 + d]) / (float)L;
  __syncthreads();
  for (int i = threadIdx.x; i < L * 64; i += blockDim.x) {
    int l = i >> 6, dd = i & 63;
    ctx[((size_t)b * L + l) * DM + h * DH + dd] = meanv[dd];
  }
}

__global__ void ctx_cumsum_kernel(const float* __restrict__ V,
                                  float* __restrict__ ctx, int L) {
  int bh = blockIdx.x; int b = bh >> 3, h = bh & 7;
  int d = threadIdx.x;  // 64 threads
  float s = 0.f;
  for (int l = 0; l < L; ++l) {
    s += V[((size_t)b * L + l) * DM + h * DH + d];
    ctx[((size_t)b * L + l) * DM + h * DH + d] = s;
  }
}

// ------------- attention core: softmax(QK^T)V for selected rows ------------
__global__ void topq_attn_kernel(const float* __restrict__ Q,
                                 const float* __restrict__ K,
                                 const float* __restrict__ V,
                                 const int* __restrict__ index,
                                 float* __restrict__ ctx,
                                 int L, int Lk, int u, int causal) {
  int gid = blockIdx.x;          // (b*NH + h)*u + ui
  int ui = gid % u;
  int bh = gid / u;
  int b = bh >> 3, h = bh & 7;
  int lsel = index ? index[bh * 64 + ui] : ui;

  __shared__ float qsh[DH];
  __shared__ float sc[2048];
  __shared__ float red[256];
  if (threadIdx.x < DH)
    qsh[threadIdx.x] = Q[((size_t)b * L + lsel) * DM + h * DH + threadIdx.x];
  __syncthreads();

  const float scale = 0.125f;    // 1/sqrt(64)
  for (int k = threadIdx.x; k < Lk; k += blockDim.x) {
    const float* kr = K + ((size_t)b * Lk + k) * DM + h * DH;
    float d = 0.f;
#pragma unroll
    for (int j = 0; j < DH; ++j) d += qsh[j] * kr[j];
    d *= scale;
    if (causal && k > lsel) d = NEG_INF_F;
    sc[k] = d;
  }
  __syncthreads();

  float mx = NEG_INF_F;
  for (int k = threadIdx.x; k < Lk; k += blockDim.x) mx = fmaxf(mx, sc[k]);
  red[threadIdx.x] = mx; __syncthreads();
  for (int s = 128; s >= 1; s >>= 1) {
    if (threadIdx.x < s)
      red[threadIdx.x] = fmaxf(red[threadIdx.x], red[threadIdx.x + s]);
    __syncthreads();
  }
  mx = red[0]; __syncthreads();

  float se = 0.f;
  for (int k = threadIdx.x; k < Lk; k += blockDim.x) {
    float e = __expf(sc[k] - mx);
    sc[k] = e; se += e;
  }
  red[threadIdx.x] = se; __syncthreads();
  for (int s = 128; s >= 1; s >>= 1) {
    if (threadIdx.x < s) red[threadIdx.x] += red[threadIdx.x + s];
    __syncthreads();
  }
  se = red[0]; __syncthreads();

  int d = threadIdx.x & 63, seg = threadIdx.x >> 6;
  float acc = 0.f;
  for (int k = seg; k < Lk; k += 4)
    acc += sc[k] * V[((size_t)b * Lk + k) * DM + h * DH + d];
  red[threadIdx.x] = acc; __syncthreads();
  if (seg == 0) {
    float o = (red[d] + red[64 + d] + red[128 + d] + red[192 + d]) / se;
    ctx[((size_t)b * L + lsel) * DM + h * DH + d] = o;
  }
}

// --------------------------- final projection ------------------------------
__global__ void proj_kernel(const float* __restrict__ X,
                            const float* __restrict__ w,
                            const float* __restrict__ bias,
                            float* __restrict__ out) {
  int row = blockIdx.x;          // b*512 + t
  int b = row >> 9, t = row & 511;
  int wave = threadIdx.x >> 5, lane = threadIdx.x & 31;
  if (wave >= 7) return;
  const float* xr = X + ((size_t)b * LDEC + 512 + t) * DM;
  float s = 0.f;
  for (int j = lane; j < DM; j += 32) s += xr[j] * w[(size_t)j * 7 + wave];
#pragma unroll
  for (int m = 16; m >= 1; m >>= 1) s += __shfl_xor(s, m, 32);
  if (lane == 0) out[((size_t)b * 512 + t) * 7 + wave] = s + bias[wave];
}

// ---------------------------------------------------------------------------

struct LinP  { const float* w; const float* b; };
struct AttnP { LinP q, k, v, o; };
struct NormP { const float* g; const float* b; };
struct FfnP  { LinP w1, w2; };

extern "C" void kernel_launch(void* const* d_in, const int* in_sizes, int n_in,
                              void* d_out, int out_size, void* d_ws, size_t ws_size,
                              hipStream_t stream) {
  const float* x_enc      = (const float*)d_in[0];
  const int*   x_mark_enc = (const int*)d_in[1];
  const float* x_dec      = (const float*)d_in[2];
  const int*   x_mark_dec = (const int*)d_in[3];

  // Params flattened in JAX pytree order (dict keys sorted):
  // dec_conv, dec_layers[0..1]{cross(k,o,q,v each b,w), ffn(w1 b,w; w2 b,w),
  // n1(b,g), n2(b,g), n3(b,g), self(k,o,q,v each b,w)}, dec_norm(b,g),
  // enc_conv, enc_layers[0..2]{attn, ffn, n1, n2}, enc_norm(b,g), proj(b,w)
  int cur = 4;
  auto nf = [&]() { return (const float*)d_in[cur++]; };
  auto rdLin  = [&](LinP& l)  { l.b = nf(); l.w = nf(); };
  auto rdAttn = [&](AttnP& a) { rdLin(a.k); rdLin(a.o); rdLin(a.q); rdLin(a.v); };
  auto rdNorm = [&](NormP& n) { n.b = nf(); n.g = nf(); };
  auto rdFfn  = [&](FfnP& f)  { rdLin(f.w1); rdLin(f.w2); };

  const float* dec_conv = nf();
  AttnP dcross[2], dself[2]; FfnP dffn[2]; NormP dn1[2], dn2[2], dn3[2];
  for (int i = 0; i < 2; ++i) {
    rdAttn(dcross[i]); rdFfn(dffn[i]);
    rdNorm(dn1[i]); rdNorm(dn2[i]); rdNorm(dn3[i]);
    rdAttn(dself[i]);
  }
  NormP dec_norm; rdNorm(dec_norm);
  const float* enc_conv = nf();
  AttnP eattn[3]; FfnP effn[3]; NormP en1[3], en2[3];
  for (int i = 0; i < 3; ++i) {
    rdAttn(eattn[i]); rdFfn(effn[i]); rdNorm(en1[i]); rdNorm(en2[i]);
  }
  NormP enc_norm; rdNorm(enc_norm);
  LinP proj; rdLin(proj);

  // workspace carve
  float* ws = (float*)d_ws;
  const size_t U = (size_t)8192 * DM;     // 16 MB slabs
  float* bEnc = ws + 0 * U;
  float* bQ   = ws + 1 * U;
  float* bK   = ws + 2 * U;
  float* bV   = ws + 3 * U;
  float* bCtx = ws + 4 * U;
  float* bX   = ws + 5 * U;
  float* bT   = ws + 6 * U;
  float* bDec = ws + 7 * U;
  float* bM   = ws + 8 * U;               // 32*2048 floats
  int*   bIdx = (int*)(ws + 8 * U + 32 * 2048);

  const int ME = BB * LSEQ;   // 8192
  const int MD = BB * LDEC;   // 4096

  auto gemm = [&](const float* A, const LinP& l, float* C, int M, int act) {
    dim3 g(M / BM, DM / BN);
    gemm_bias_kernel<<<g, 256, 0, stream>>>(A, l.w, l.b, C, M, DM, DM, act);
  };
  auto addln = [&](const float* x, const float* y, const NormP& n, float* o, int M) {
    add_ln_kernel<<<M / 8, 256, 0, stream>>>(x, y, n.g, n.b, o, M);
  };

  // ------------------------------ encoder ---------------------------------
  embed_kernel<<<ME, 256, 0, stream>>>(x_enc, x_mark_enc, enc_conv, bEnc, LSEQ);
  for (int i = 0; i < 3; ++i) {
    gemm(bEnc, eattn[i].q, bQ, ME, 0);
    gemm(bEnc, eattn[i].k, bK, ME, 0);
    gemm(bEnc, eattn[i].v, bV, ME, 0);
    prob_sample_m_kernel<<<(BB * NH * LSEQ) / 8, 256, 0, stream>>>(
        bQ, bK, bM, LSEQ, LSEQ, 40, (unsigned)i, BB * NH * LSEQ);
    topk_kernel<<<BB * NH, 256, 0, stream>>>(bM, bIdx, LSEQ, 40);
    ctx_mean_kernel<<<BB * NH, 256, 0, stream>>>(bV, bCtx, LSEQ);
    topq_attn_kernel<<<BB * NH * 40, 256, 0, stream>>>(
        bQ, bK, bV, bIdx, bCtx, LSEQ, LSEQ, 40, 0);
    gemm(bCtx, eattn[i].o, bT, ME, 0);
    addln(bEnc, bT, en1[i], bX, ME);
    gemm(bX, effn[i].w1, bT, ME, 1);      // GELU
    gemm(bT, effn[i].w2, bQ, ME, 0);
    addln(bX, bQ, en2[i], bEnc, ME);
  }
  addln(bEnc, nullptr, enc_norm, bEnc, ME);

  // ------------------------------ decoder ---------------------------------
  embed_kernel<<<MD, 256, 0, stream>>>(x_dec, x_mark_dec, dec_conv, bDec, LDEC);
  for (int i = 0; i < 2; ++i) {
    // causal ProbSparse self-attention
    gemm(bDec, dself[i].q, bQ, MD, 0);
    gemm(bDec, dself[i].k, bK, MD, 0);
    gemm(bDec, dself[i].v, bV, MD, 0);
    prob_sample_m_kernel<<<(BB * NH * LDEC) / 8, 256, 0, stream>>>(
        bQ, bK, bM, LDEC, LDEC, 35, (unsigned)(100 + i), BB * NH * LDEC);
    topk_kernel<<<BB * NH, 256, 0, stream>>>(bM, bIdx, LDEC, 35);
    ctx_cumsum_kernel<<<BB * NH, 64, 0, stream>>>(bV, bCtx, LDEC);
    topq_attn_kernel<<<BB * NH * 35, 256, 0, stream>>>(
        bQ, bK, bV, bIdx, bCtx, LDEC, LDEC, 35, 1);
    gemm(bCtx, dself[i].o, bT, MD, 0);
    addln(bDec, bT, dn1[i], bX, MD);
    // full cross-attention against encoder output
    gemm(bX,   dcross[i].q, bQ, MD, 0);
    gemm(bEnc, dcross[i].k, bK, ME, 0);
    gemm(bEnc, dcross[i].v, bV, ME, 0);
    topq_attn_kernel<<<BB * NH * LDEC, 256, 0, stream>>>(
        bQ, bK, bV, nullptr, bCtx, LDEC, LSEQ, LDEC, 0);
    gemm(bCtx, dcross[i].o, bT, MD, 0);
    addln(bX, bT, dn2[i], bCtx, MD);      // x2
    // FFN
    gemm(bCtx, dffn[i].w1, bT, MD, 1);    // GELU
    gemm(bT, dffn[i].w2, bQ, MD, 0);
    addln(bCtx, bQ, dn3[i], bDec, MD);
  }
  addln(bDec, nullptr, dec_norm, bX, MD);
  proj_kernel<<<BB * 512, 256, 0, stream>>>(bX, proj.w, proj.b, (float*)d_out);
}
